// GATEncoder_17076789969562
// MI455X (gfx1250) — matrix operand order repaired and checked by reading, not compiled, not run on hardware
//
#include <hip/hip_runtime.h>
#include <hip/hip_bf16.h>
#include <cstdint>

#define BB 16
#define NN 512
#define MM 16
#define HH 8
#define DD 128
#define ROWS (BB*NN)   // 8192

typedef __attribute__((ext_vector_type(16))) __bf16 v16bf;
typedef __attribute__((ext_vector_type(8)))  __bf16 v8bf;
typedef __attribute__((ext_vector_type(4)))  __bf16 v4bf;
typedef __attribute__((ext_vector_type(8)))  float  v8f;

union V16 { v16bf v; v8bf h[2]; };

#if __has_builtin(__builtin_amdgcn_s_wait_asynccnt)
#define WAIT_ASYNC0() __builtin_amdgcn_s_wait_asynccnt(0)
#else
#define WAIT_ASYNC0() asm volatile("s_wait_asynccnt 0" ::: "memory")
#endif

// ---------------- fp32 -> bf16 elementwise ----------------
__global__ void cvt_kernel(const float* __restrict__ src, __bf16* __restrict__ dst, int n) {
  int i = blockIdx.x * blockDim.x + threadIdx.x;
  if (i < n) dst[i] = (__bf16)src[i];
}

// ---------------- batched transpose + convert: dst[b][c][r] = src[b][r][c] ----------------
__global__ void cvtT_kernel(const float* __restrict__ src, __bf16* __restrict__ dst,
                            int batch, int R, int C) {
  int i = blockIdx.x * blockDim.x + threadIdx.x;
  int total = batch * R * C;
  if (i >= total) return;
  int b  = i / (R * C);
  int rc = i - b * R * C;
  int r  = rc / C;
  int c  = rc - r * C;
  dst[(size_t)b * R * C + (size_t)c * R + r] = (__bf16)src[i];
}

// ---------------- bf16 WMMA GEMM ----------------
// block = 16-row stripe, wave = one 16-col tile. A staged to LDS in 16x128 chunks via
// cooperative async global->LDS b128 copies (ASYNCcnt), double-buffered; one barrier
// covers 4 unrolled WMMAs. B fragments from global (column-major Bt[N][K]).
template <int KTOT>
__global__ void gemm_bf16_kernel(const __bf16* __restrict__ A, int lda,
                                 const __bf16* __restrict__ Bt,
                                 float* __restrict__ outF, __bf16* __restrict__ outB, int ldc,
                                 const float* __restrict__ bias,
                                 const float* __restrict__ resid, int ldr,
                                 int relu)
{
  constexpr int KC = 128;                 // staged K-chunk
  constexpr int SH_LDA = KC + 8;          // +16B row pad: de-conflict LDS banks
  __shared__ __align__(16) __bf16 shA[2][16 * SH_LDA];   // 2 x ~4.25KB

  int tid  = threadIdx.x;
  int wave = tid >> 5;
  int lane = tid & 31;
  int r0 = blockIdx.x << 4;               // row stripe
  int c0 = wave << 4;                     // column tile = wave id
  int half = lane >> 4;
  int l15  = lane & 15;

  const char* aPanel   = (const char*)(A + (size_t)r0 * lda);
  const __bf16* bRow   = Bt + (size_t)(c0 + l15) * KTOT;

  // stage one 16x128 bf16 chunk: 256 x 16B pieces spread over the block
  auto stage = [&](int kc0, int sbuf) {
    for (int p = tid; p < 256; p += blockDim.x) {
      int row  = p >> 4;                  // 0..15
      int boff = (p & 15) * 16;           // byte offset within 256B row-chunk
      uint64_t g = (uint64_t)(uintptr_t)(aPanel + ((size_t)row * lda + kc0) * 2 + boff);
      uint32_t l = (uint32_t)(uintptr_t)((char*)shA[sbuf] + row * (SH_LDA * 2) + boff);
      asm volatile("global_load_async_to_lds_b128 %0, %1, off" :: "v"(l), "v"(g) : "memory");
    }
  };

  stage(0, 0);
  v8f acc = {};
  int buf = 0;
#pragma unroll 1
  for (int kc0 = 0; kc0 < KTOT; kc0 += KC) {
    WAIT_ASYNC0();                        // each wave's own async pieces landed
    __syncthreads();                      // publish chunk to all waves
    if (kc0 + KC < KTOT) stage(kc0 + KC, buf ^ 1);   // prefetch next chunk

    const __bf16* aC = shA[buf] + l15 * SH_LDA;
#pragma unroll
    for (int k = 0; k < KC; k += 32) {
      V16 a, b;
      // A 16x32 fragment from LDS: lane<16 -> K {0..7,16..23}; lane>=16 -> K {8..15,24..31}
      a.h[0] = *(const v8bf*)(aC + k + half * 8);
      a.h[1] = *(const v8bf*)(aC + k + 16 + half * 8);
      // B 32x16 fragment (column-major): lane<16 -> K {0..7,16..23}; lane>=16 -> K {8..15,24..31}
      b.h[0] = *(const v8bf*)(bRow + kc0 + k + half * 8);
      b.h[1] = *(const v8bf*)(bRow + kc0 + k + 16 + half * 8);
      acc = __builtin_amdgcn_wmma_f32_16x16x32_bf16(false, a.v, false, b.v,
                                                    (short)0, acc, false, false);
    }
    buf ^= 1;
  }

  int col = c0 + l15;
  float bv = bias ? bias[col] : 0.0f;
#pragma unroll
  for (int i = 0; i < 8; ++i) {
    int row = r0 + i + half * 8;          // C/D layout: VGPR i -> row i (+8 for upper half)
    float v = acc[i] + bv;
    if (resid) v += resid[(size_t)row * ldr + col];
    if (relu)  v = v > 0.0f ? v : 0.0f;
    if (outF)  outF[(size_t)row * ldc + col] = v;
    if (outB)  outB[(size_t)row * ldc + col] = (__bf16)v;
  }
}

// ---------------- s_src/s_dst = Wh . a_src / a_dst (wave per (bn,h)) ----------------
__global__ void score_kernel(const __bf16* __restrict__ Wh,
                             const float* __restrict__ a_src,
                             const float* __restrict__ a_dst,
                             float* __restrict__ s_src, float* __restrict__ s_dst)
{
  int wave = threadIdx.x >> 5, lane = threadIdx.x & 31;
  int w = blockIdx.x * (blockDim.x >> 5) + wave;
  int bn = w >> 3, h = w & 7;
  if (bn >= ROWS) return;
  const __bf16* row = Wh + (size_t)bn * HH * DD + h * DD;
  const float* as = a_src + h * DD;
  const float* ad = a_dst + h * DD;
  float s0 = 0.0f, s1 = 0.0f;
#pragma unroll
  for (int j = 0; j < 4; ++j) {
    int e = lane + 32 * j;
    float x = (float)row[e];
    s0 += x * as[e];
    s1 += x * ad[e];
  }
  for (int o = 16; o >= 1; o >>= 1) { s0 += __shfl_xor(s0, o, 32); s1 += __shfl_xor(s1, o, 32); }
  if (lane == 0) { s_src[w] = s0; s_dst[w] = s1; }
}

// ---------------- attention + aggregation (block per node, wave per head) ----------------
__global__ void attn_kernel(const __bf16* __restrict__ Wh,
                            const int* __restrict__ adj, const int* __restrict__ n_list,
                            const float* __restrict__ s_src, const float* __restrict__ s_dst,
                            __bf16* __restrict__ att)
{
  __shared__ int sh_nb[MM];
  __shared__ int sh_adj[MM];
  int bn = blockIdx.x;
  int b  = bn >> 9;                             // N = 512
  int tid = threadIdx.x;
  if (tid < MM)            sh_nb[tid]        = n_list[(size_t)bn * MM + tid];
  else if (tid < 2 * MM)   sh_adj[tid - MM]  = adj[(size_t)bn * MM + (tid - MM)];
  __syncthreads();

  int h = tid >> 5, lane = tid & 31;
  int m = lane & 15;                            // both halves replicate the 16 scores
  int nbm = sh_nb[m];
  float sc = s_src[bn * HH + h] + s_dst[((size_t)(b * NN + nbm)) * HH + h];
  sc = sc > 0.0f ? sc : 0.2f * sc;              // leaky relu
  if (sh_adj[m] <= 0) sc = -1e9f;

  float mx = sc;
  for (int o = 8; o >= 1; o >>= 1) mx = fmaxf(mx, __shfl_xor(mx, o, 16));
  float ex = __expf(sc - mx);
  float sm = ex;
  for (int o = 8; o >= 1; o >>= 1) sm += __shfl_xor(sm, o, 16);
  float attn = ex / sm;

  // out[e0..e0+3] = sum_m attn_m * Wh[neigh_m, h, e]  (vectorized b64 gathers)
  int e0 = lane * 4;
  float a0 = 0.0f, a1 = 0.0f, a2 = 0.0f, a3 = 0.0f;
#pragma unroll
  for (int mm2 = 0; mm2 < MM; ++mm2) {
    float wgt = __shfl(attn, mm2, 32);          // attn for neighbor mm2 lives in lane mm2
    int nb = sh_nb[mm2];
    const v4bf w = *(const v4bf*)(Wh + ((size_t)(b * NN + nb)) * HH * DD + h * DD + e0);
    a0 += wgt * (float)w[0];
    a1 += wgt * (float)w[1];
    a2 += wgt * (float)w[2];
    a3 += wgt * (float)w[3];
  }
  v4bf o;
  o[0] = (__bf16)a0; o[1] = (__bf16)a1; o[2] = (__bf16)a2; o[3] = (__bf16)a3;
  *(v4bf*)(att + (size_t)bn * HH * DD + h * DD + e0) = o;
}

// ---------------- layernorm, wave per row (D = 128 -> 4 elems/lane) ----------------
__global__ void ln_kernel(const float* __restrict__ x, const float* __restrict__ g,
                          const float* __restrict__ bta, float* __restrict__ outF,
                          __bf16* __restrict__ outB, int rows)
{
  int wave = threadIdx.x >> 5, lane = threadIdx.x & 31;
  int row = blockIdx.x * (blockDim.x >> 5) + wave;
  if (row >= rows) return;
  const float* xr = x + (size_t)row * DD;
  float v[4]; float s = 0.0f;
#pragma unroll
  for (int j = 0; j < 4; ++j) { v[j] = xr[lane + 32 * j]; s += v[j]; }
  for (int o = 16; o >= 1; o >>= 1) s += __shfl_xor(s, o, 32);
  float mu = s * (1.0f / DD);
  float q = 0.0f;
#pragma unroll
  for (int j = 0; j < 4; ++j) { float d = v[j] - mu; q += d * d; }
  for (int o = 16; o >= 1; o >>= 1) q += __shfl_xor(q, o, 32);
  float rs = rsqrtf(q * (1.0f / DD) + 1e-5f);
#pragma unroll
  for (int j = 0; j < 4; ++j) {
    int e = lane + 32 * j;
    float y = (v[j] - mu) * rs * g[e] + bta[e];
    if (outF) outF[(size_t)row * DD + e] = y;
    if (outB) outB[(size_t)row * DD + e] = (__bf16)y;
  }
}

extern "C" void kernel_launch(void* const* d_in, const int* in_sizes, int n_in,
                              void* d_out, int out_size, void* d_ws, size_t ws_size,
                              hipStream_t stream) {
  const float* h     = (const float*)d_in[0];
  const int*   adj   = (const int*)d_in[1];
  const int*   nlist = (const int*)d_in[2];
  const float* W     = (const float*)d_in[3];
  const float* a_src = (const float*)d_in[4];
  const float* a_dst = (const float*)d_in[5];
  const float* Wo    = (const float*)d_in[6];
  const float* ln1g  = (const float*)d_in[7];
  const float* ln1b  = (const float*)d_in[8];
  const float* w1    = (const float*)d_in[9];
  const float* b1    = (const float*)d_in[10];
  const float* w2    = (const float*)d_in[11];
  const float* b2    = (const float*)d_in[12];
  const float* ln2g  = (const float*)d_in[13];
  const float* ln2b  = (const float*)d_in[14];

  char* p = (char*)d_ws;
  auto take = [&](size_t bytes) -> char* {
    uintptr_t q = ((uintptr_t)p + 255) & ~(uintptr_t)255;
    p = (char*)q + bytes;
    return (char*)q;
  };
  __bf16* hbf  = (__bf16*)take((size_t)ROWS * DD * 2);          // h in bf16
  __bf16* Wt   = (__bf16*)take((size_t)HH * DD * DD * 2);       // W^T per head [h][e][d]
  __bf16* WoT  = (__bf16*)take((size_t)DD * (HH * DD) * 2);     // Wo^T [128][1024]
  __bf16* w1T  = (__bf16*)take((size_t)(2 * DD) * DD * 2);      // ff_w1^T [256][128]
  __bf16* w2T  = (__bf16*)take((size_t)DD * (2 * DD) * 2);      // ff_w2^T [128][256]
  __bf16* Whbf = (__bf16*)take((size_t)ROWS * HH * DD * 2);     // Wh [8192][1024]
  float*  ssrc = (float*)take((size_t)ROWS * HH * 4);
  float*  sdst = (float*)take((size_t)ROWS * HH * 4);
  __bf16* attb = (__bf16*)take((size_t)ROWS * HH * DD * 2);     // concat-heads attn out
  float*  buf1 = (float*)take((size_t)ROWS * DD * 4);           // Wo out + residual (pre-LN1)
  float*  tF   = (float*)take((size_t)ROWS * DD * 4);           // t fp32
  __bf16* tB   = (__bf16*)take((size_t)ROWS * DD * 2);          // t bf16
  __bf16* uB   = (__bf16*)take((size_t)ROWS * 2 * DD * 2);      // FFN hidden bf16
  float*  vF   = (float*)take((size_t)ROWS * DD * 4);           // pre-LN2
  (void)ws_size; (void)in_sizes; (void)n_in; (void)out_size;

  // --- conversions / weight transposes ---
  int nH = ROWS * DD;
  cvt_kernel<<<(nH + 255) / 256, 256, 0, stream>>>(h, hbf, nH);
  cvtT_kernel<<<(HH * DD * DD + 255) / 256, 256, 0, stream>>>(W,  Wt,  HH, DD, DD);
  cvtT_kernel<<<(HH * DD * DD + 255) / 256, 256, 0, stream>>>(Wo, WoT, 1, HH * DD, DD);
  cvtT_kernel<<<(DD * 2 * DD + 255) / 256, 256, 0, stream>>>(w1, w1T, 1, DD, 2 * DD);
  cvtT_kernel<<<(DD * 2 * DD + 255) / 256, 256, 0, stream>>>(w2, w2T, 1, 2 * DD, DD);

  // --- per-head projection: Wh[:,h,:] = h @ W[h]  (WMMA, single async-staged K chunk) ---
  for (int hh = 0; hh < HH; ++hh) {
    gemm_bf16_kernel<DD><<<ROWS / 16, 256, 0, stream>>>(   // 8 waves = 8 column tiles
        hbf, DD, Wt + (size_t)hh * DD * DD,
        nullptr, Whbf + hh * DD, HH * DD,
        nullptr, nullptr, 0, 0);
  }

  // --- attention scores + softmax aggregation ---
  score_kernel<<<ROWS * HH / 8, 256, 0, stream>>>(Whbf, a_src, a_dst, ssrc, sdst);
  attn_kernel<<<ROWS, 256, 0, stream>>>(Whbf, adj, nlist, ssrc, sdst, attb);

  // --- output projection + residual h (WMMA, K=1024, 8 double-buffered chunks) ---
  gemm_bf16_kernel<HH * DD><<<ROWS / 16, 256, 0, stream>>>(
      attb, HH * DD, WoT, buf1, nullptr, DD,
      nullptr, h, DD, 0);
  ln_kernel<<<ROWS / 8, 256, 0, stream>>>(buf1, ln1g, ln1b, tF, tB, ROWS);

  // --- FFN1: relu(t @ w1 + b1) (WMMA, 16 column tiles -> 512 threads) ---
  gemm_bf16_kernel<DD><<<ROWS / 16, 512, 0, stream>>>(
      tB, DD, w1T, nullptr, uB, 2 * DD,
      b1, nullptr, 0, 1);

  // --- FFN2: u @ w2 + b2 + t (WMMA, K=256, 2 chunks) ---
  gemm_bf16_kernel<2 * DD><<<ROWS / 16, 256, 0, stream>>>(
      uB, 2 * DD, w2T, vF, nullptr, DD,
      b2, tF, DD, 0);

  ln_kernel<<<ROWS / 8, 256, 0, stream>>>(vF, ln2g, ln2b, (float*)d_out, nullptr, ROWS);
}
